// AttentionModule_55095840473245
// MI455X (gfx1250) — matrix operand order repaired and checked by reading, not compiled, not run on hardware
//
#include <hip/hip_runtime.h>
#include <hip/hip_bf16.h>
#include <math.h>
#include <stdint.h>

// ---------------------------------------------------------------------------
// Problem constants (from reference setup_inputs)
// ---------------------------------------------------------------------------
#define BB 8
#define SS 1024
#define DD 1024
#define NN_ 1024
#define CC 32
#define KC 4
#define HH 16
#define DHH 64   // DD / HH

typedef __bf16 bf16_t;
typedef __attribute__((ext_vector_type(16))) __bf16 v16bf;
typedef __attribute__((ext_vector_type(8)))  __bf16 v8bf;
typedef __attribute__((ext_vector_type(8)))  float  v8f;

// ---------------------------------------------------------------------------
// WMMA fragment helpers (wave32, bf16 16x16x32, per CDNA5 ISA layouts)
// A 16x32 (MxK): lanes0-15 row M=l, K0..7 in V0-3, K16..23 in V4-7;
//                lanes16-31 same rows, K8..15 / K24..31.
// B 32x16 (KxN): lane n (0-15) holds col N=n with K0..7, K16..23; lanes16-31 K8..15 / K24..31.
// NT convention: both operands come from row-major [rows x Kdim] storage.
// ---------------------------------------------------------------------------
__device__ inline v16bf load_frag_a(const bf16_t* src, int ld, int k0) {
  int l = threadIdx.x & 31;
  int r = l & 15;
  int h = (l >> 4) & 1;
  const bf16_t* p = src + (size_t)r * ld + k0 + h * 8;
  v8bf lo = *(const v8bf*)p;
  v8bf hi = *(const v8bf*)(p + 16);
  v16bf out;
#pragma unroll
  for (int i = 0; i < 8; ++i) { out[i] = lo[i]; out[i + 8] = hi[i]; }
  return out;
}

__device__ inline v16bf load_frag_b(const bf16_t* src, int ld, int k0) {
  int l = threadIdx.x & 31;
  int n = l & 15;
  int h = (l >> 4) & 1;
  const bf16_t* p = src + (size_t)n * ld + k0 + h * 8;
  v8bf lo = *(const v8bf*)p;
  v8bf hi = *(const v8bf*)(p + 16);
  v16bf out;
#pragma unroll
  for (int i = 0; i < 8; ++i) { out[i] = lo[i]; out[i + 8] = hi[i]; }
  return out;
}

__device__ inline v8f wmma_bf16(v16bf a, v16bf b, v8f c) {
  return __builtin_amdgcn_wmma_f32_16x16x32_bf16(false, a, false, b, (short)0, c,
                                                 false, false);
}

// Async global -> LDS copy, 16B per lane (tracked by ASYNCcnt).
// GVS mode: 64-bit SGPR base + per-lane 32-bit byte offset.
__device__ inline void async_cp_b128(uint32_t lds_off, uint32_t glb_off,
                                     const bf16_t* base) {
  asm volatile("global_load_async_to_lds_b128 %0, %1, %2"
               :
               : "v"(lds_off), "v"(glb_off), "s"(base)
               : "memory");
}

// ---------------------------------------------------------------------------
// f32 -> bf16 conversion
// ---------------------------------------------------------------------------
__global__ __launch_bounds__(256) void cvt_bf16(const float* __restrict__ in,
                                                bf16_t* __restrict__ out, size_t n) {
  size_t i = (size_t)blockIdx.x * blockDim.x + threadIdx.x;
  if (i < n) out[i] = (bf16_t)in[i];
}

// ---------------------------------------------------------------------------
// Circuit weight build:
//   W[b,n,d] = sum_k w[b,k] * inner[b,k,n] * table[idx[b,k], n, d]
// trans=1 stores W^T as [b, d, n] (so downstream GEMMs are all NT).
// One block per (b, n).
// ---------------------------------------------------------------------------
__global__ __launch_bounds__(256) void build_w(const float* __restrict__ table,
                                               const int* __restrict__ idx,
                                               const float* __restrict__ inner,
                                               const float* __restrict__ w,
                                               bf16_t* __restrict__ out, int trans) {
  int bn = blockIdx.x;
  int b = bn / NN_;
  int n = bn % NN_;
  float coef[KC];
  const float* tp[KC];
#pragma unroll
  for (int k = 0; k < KC; ++k) {
    int c = idx[b * KC + k];
    coef[k] = w[b * KC + k] * inner[(size_t)(b * KC + k) * NN_ + n];
    tp[k] = table + ((size_t)c * NN_ + n) * DD;
  }
  for (int d = threadIdx.x; d < DD; d += 256) {
    float s = 0.f;
#pragma unroll
    for (int k = 0; k < KC; ++k) s += coef[k] * tp[k][d];
    size_t o = trans ? ((size_t)b * DD * NN_ + (size_t)d * NN_ + n)
                     : ((size_t)b * NN_ * DD + (size_t)n * DD + d);
    out[o] = (bf16_t)s;
  }
}

// ---------------------------------------------------------------------------
// Batched NT GEMM: C[b] = A[b] (MxK, row-major) * B[b] (NxK, row-major)^T
// Block: 256 threads = 8 waves; block tile 128x128; wave tile 32x64.
// K-panels (128x32 bf16 each for A and B) are staged in LDS via
// GLOBAL_LOAD_ASYNC_TO_LDS_B128 with double buffering: while the 8 WMMAs of
// step i run from LDS buffer i&1, the async engine fills buffer (i+1)&1.
// Rows padded to 80B (TPAD=40 elems) to spread LDS banks.
// TRANS_OUT stores C transposed ([n, m], ldc = M); F32_OUT stores float.
// ---------------------------------------------------------------------------
#define TPAD 40

template <bool TRANS_OUT, bool F32_OUT>
__global__ __launch_bounds__(256) void gemm_nt(const bf16_t* __restrict__ A,
                                               size_t strideA, int lda,
                                               const bf16_t* __restrict__ B,
                                               size_t strideB, int ldb,
                                               void* __restrict__ Cout,
                                               size_t strideC, int ldc, int Kdim) {
  __shared__ __align__(16) bf16_t smA[2][128 * TPAD];
  __shared__ __align__(16) bf16_t smB[2][128 * TPAD];

  int bz = blockIdx.z;
  const bf16_t* Ab = A + (size_t)bz * strideA + (size_t)(blockIdx.x * 128) * lda;
  const bf16_t* Bb = B + (size_t)bz * strideB + (size_t)(blockIdx.y * 128) * ldb;

  int wave = threadIdx.x >> 5;
  int lane = threadIdx.x & 31;
  int wm = wave & 3;   // 4 waves along M
  int wn = wave >> 2;  // 2 waves along N

  // Async-copy mapping: panel = 128 rows x 4 chunks of 16B -> 512 chunks.
  // Each wave issues chunks {wave*64+lane, wave*64+32+lane} per panel.
  int c0 = wave * 64 + lane;
  int c1 = c0 + 32;
  int r0 = c0 >> 2, kc0 = c0 & 3;
  int r1 = c1 >> 2, kc1 = c1 & 3;
  uint32_t gA0 = (uint32_t)((r0 * lda + kc0 * 8) * 2);
  uint32_t gA1 = (uint32_t)((r1 * lda + kc1 * 8) * 2);
  uint32_t gB0 = (uint32_t)((r0 * ldb + kc0 * 8) * 2);
  uint32_t gB1 = (uint32_t)((r1 * ldb + kc1 * 8) * 2);
  uint32_t lo0 = (uint32_t)(r0 * (TPAD * 2) + kc0 * 16);
  uint32_t lo1 = (uint32_t)(r1 * (TPAD * 2) + kc1 * 16);
  uint32_t ldsA0 = (uint32_t)(uint64_t)(uintptr_t)&smA[0][0];
  uint32_t ldsA1 = (uint32_t)(uint64_t)(uintptr_t)&smA[1][0];
  uint32_t ldsB0 = (uint32_t)(uint64_t)(uintptr_t)&smB[0][0];
  uint32_t ldsB1 = (uint32_t)(uint64_t)(uintptr_t)&smB[1][0];

  auto issue = [&](int buf, int k0) {
    const bf16_t* a = Ab + k0;
    const bf16_t* b = Bb + k0;
    uint32_t la = buf ? ldsA1 : ldsA0;
    uint32_t lb = buf ? ldsB1 : ldsB0;
    async_cp_b128(la + lo0, gA0, a);
    async_cp_b128(la + lo1, gA1, a);
    async_cp_b128(lb + lo0, gB0, b);
    async_cp_b128(lb + lo1, gB1, b);
  };

  v8f acc[2][4];
#pragma unroll
  for (int i = 0; i < 2; ++i)
#pragma unroll
    for (int j = 0; j < 4; ++j) acc[i][j] = (v8f){};

  int nsteps = Kdim >> 5;
  issue(0, 0);
  for (int i = 0; i < nsteps; ++i) {
    if (i + 1 < nsteps) {
      issue((i + 1) & 1, (i + 1) << 5);
      // 8 outstanding; <=4 left means the current panel's batch retired.
      asm volatile("s_wait_asynccnt 0x4" ::: "memory");
    } else {
      asm volatile("s_wait_asynccnt 0x0" ::: "memory");
    }
    __syncthreads();  // current panel visible to all waves

    int buf = i & 1;
    v16bf a0 = load_frag_a(&smA[buf][(wm * 32) * TPAD], TPAD, 0);
    v16bf a1 = load_frag_a(&smA[buf][(wm * 32 + 16) * TPAD], TPAD, 0);
    v16bf bf[4];
#pragma unroll
    for (int j = 0; j < 4; ++j)
      bf[j] = load_frag_b(&smB[buf][(wn * 64 + 16 * j) * TPAD], TPAD, 0);
#pragma unroll
    for (int j = 0; j < 4; ++j) {
      acc[0][j] = wmma_bf16(a0, bf[j], acc[0][j]);
      acc[1][j] = wmma_bf16(a1, bf[j], acc[1][j]);
    }
    __syncthreads();  // everyone done with this buffer before it is refilled
  }

  // Store: C/D layout -> lane l holds (m = e + 8*(l>>4), n = l&15) in VGPR e.
  int m0 = blockIdx.x * 128 + wm * 32;
  int n0 = blockIdx.y * 128 + wn * 64;
  int nn = lane & 15;
  int hh = (lane >> 4) & 1;
#pragma unroll
  for (int i = 0; i < 2; ++i)
#pragma unroll
    for (int j = 0; j < 4; ++j)
#pragma unroll
      for (int e = 0; e < 8; ++e) {
        int m = m0 + 16 * i + e + 8 * hh;
        int n = n0 + 16 * j + nn;
        size_t idx = TRANS_OUT ? ((size_t)n * ldc + m) : ((size_t)m * ldc + n);
        if (F32_OUT)
          ((float*)Cout)[(size_t)bz * strideC + idx] = acc[i][j][e];
        else
          ((bf16_t*)Cout)[(size_t)bz * strideC + idx] = (bf16_t)acc[i][j][e];
      }
}

// ---------------------------------------------------------------------------
// Causal flash attention. One wave per 16-row query tile, 4 waves/block.
// Q,K: [B,S,D] bf16 (head h at column h*64). Vt: [B, D, S] bf16 (transposed).
// Out: [B,S,D] bf16. Softmax in f32; P re-laid out C->A via per-wave LDS.
// ---------------------------------------------------------------------------
__global__ __launch_bounds__(128) void flash_attn(const bf16_t* __restrict__ Q,
                                                  const bf16_t* __restrict__ K,
                                                  const bf16_t* __restrict__ Vt,
                                                  bf16_t* __restrict__ Oa) {
  __shared__ __align__(16) bf16_t pbuf[4][16 * 32];
  int wave = threadIdx.x >> 5;
  int l = threadIdx.x & 31;
  int nn = l & 15;
  int hh = (l >> 4) & 1;

  int gid = blockIdx.x;            // over B*H*(S/64)
  int qblk = gid % (SS / 64);
  int bh = gid / (SS / 64);
  int h = bh % HH;
  int b = bh / HH;
  int q0 = qblk * 64 + wave * 16;

  const bf16_t* Qh = Q + ((size_t)b * SS + q0) * DD + h * DHH;
  const bf16_t* Kh = K + (size_t)b * SS * DD + h * DHH;
  const bf16_t* Vh = Vt + ((size_t)b * DD + h * DHH) * SS;

  v16bf qa0 = load_frag_a(Qh, DD, 0);
  v16bf qa1 = load_frag_a(Qh, DD, 32);

  v8f o[4];
#pragma unroll
  for (int nt = 0; nt < 4; ++nt) o[nt] = (v8f){};
  float mrow[8], lrow[8];
#pragma unroll
  for (int i = 0; i < 8; ++i) { mrow[i] = -3.0e38f; lrow[i] = 0.f; }
  const float scale = 0.125f;  // 1/sqrt(64)

  bf16_t* pb = pbuf[wave];

  for (int t0 = 0; t0 <= q0 + 15; t0 += 32) {
    // scores for keys [t0, t0+32)
    v8f s0 = (v8f){}, s1 = (v8f){};
    s0 = wmma_bf16(qa0, load_frag_b(Kh + (size_t)t0 * DD, DD, 0), s0);
    s0 = wmma_bf16(qa1, load_frag_b(Kh + (size_t)t0 * DD, DD, 32), s0);
    s1 = wmma_bf16(qa0, load_frag_b(Kh + (size_t)(t0 + 16) * DD, DD, 0), s1);
    s1 = wmma_bf16(qa1, load_frag_b(Kh + (size_t)(t0 + 16) * DD, DD, 32), s1);

    // scale + causal mask; online softmax update
#pragma unroll
    for (int i = 0; i < 8; ++i) {
      int q = q0 + i + 8 * hh;
      int t = t0 + nn;
      float v0 = (t <= q) ? s0[i] * scale : -3.0e38f;
      float v1 = (t + 16 <= q) ? s1[i] * scale : -3.0e38f;
      float mx = fmaxf(v0, v1);
#pragma unroll
      for (int d = 1; d < 16; d <<= 1) mx = fmaxf(mx, __shfl_xor(mx, d, 32));
      float mnew = fmaxf(mrow[i], mx);
      float fac = __expf(mrow[i] - mnew);
      mrow[i] = mnew;
      float p0 = __expf(v0 - mnew);
      float p1 = __expf(v1 - mnew);
      float rs = p0 + p1;
#pragma unroll
      for (int d = 1; d < 16; d <<= 1) rs += __shfl_xor(rs, d, 32);
      lrow[i] = lrow[i] * fac + rs;
#pragma unroll
      for (int nt = 0; nt < 4; ++nt) o[nt][i] *= fac;
      s0[i] = p0;
      s1[i] = p1;
    }

    // P (16x32) -> LDS in row-major, then reload in A-fragment layout
#pragma unroll
    for (int i = 0; i < 8; ++i) {
      int m = i + 8 * hh;
      pb[m * 32 + nn] = (bf16_t)s0[i];
      pb[m * 32 + 16 + nn] = (bf16_t)s1[i];
    }
    v16bf pa;
    {
      int r = l & 15;
      v8bf plo = *(const v8bf*)(pb + r * 32 + hh * 8);
      v8bf phi = *(const v8bf*)(pb + r * 32 + 16 + hh * 8);
#pragma unroll
      for (int i = 0; i < 8; ++i) { pa[i] = plo[i]; pa[i + 8] = phi[i]; }
    }

    // O += P * V  (B operand from V^T rows = dh columns)
#pragma unroll
    for (int nt = 0; nt < 4; ++nt)
      o[nt] = wmma_bf16(pa, load_frag_b(Vh + (size_t)(nt * 16) * SS, SS, t0), o[nt]);
  }

  // normalize and store
  bf16_t* out = Oa + ((size_t)b * SS + q0) * DD + h * DHH;
#pragma unroll
  for (int nt = 0; nt < 4; ++nt)
#pragma unroll
    for (int i = 0; i < 8; ++i) {
      int m = i + 8 * hh;
      out[(size_t)m * DD + nt * 16 + nn] = (bf16_t)(o[nt][i] / lrow[i]);
    }
}

// ---------------------------------------------------------------------------
// Host-side orchestration
// ---------------------------------------------------------------------------
extern "C" void kernel_launch(void* const* d_in, const int* in_sizes, int n_in,
                              void* d_out, int out_size, void* d_ws, size_t ws_size,
                              hipStream_t stream) {
  (void)in_sizes; (void)n_in; (void)out_size; (void)ws_size;

  const float* x        = (const float*)d_in[0];
  const float* tab_r    = (const float*)d_in[1];
  const float* tab_v    = (const float*)d_in[2];
  const float* tab_rel  = (const float*)d_in[3];
  const float* tab_val  = (const float*)d_in[4];
  const float* W_O      = (const float*)d_in[5];
  const int*   i_r      = (const int*)d_in[6];
  const float* w_r      = (const float*)d_in[7];
  const float* n_r      = (const float*)d_in[8];
  const int*   i_v      = (const int*)d_in[9];
  const float* w_v      = (const float*)d_in[10];
  const float* n_v      = (const float*)d_in[11];
  const int*   i_rq     = (const int*)d_in[12];
  const float* w_rq     = (const float*)d_in[13];
  const float* n_rq     = (const float*)d_in[14];
  const int*   i_rk     = (const int*)d_in[15];
  const float* w_rk     = (const float*)d_in[16];
  const float* n_rk     = (const float*)d_in[17];
  const int*   i_val    = (const int*)d_in[18];
  const float* w_val    = (const float*)d_in[19];
  const float* n_val    = (const float*)d_in[20];

  const size_t MAT = (size_t)BB * SS * DD;         // 8M elements
  char* ws = (char*)d_ws;
  bf16_t* xb   = (bf16_t*)(ws);                       // [B,S,D]
  bf16_t* wob  = (bf16_t*)(ws + MAT * 2);             // [D,D]
  bf16_t* bWr  = (bf16_t*)(ws + MAT * 2 + (size_t)DD * DD * 2);
  bf16_t* bWv  = bWr + MAT;                           // [B,N,D]
  bf16_t* bWtQ = bWv + MAT;                           // [B,D,N]
  bf16_t* bWtK = bWtQ + MAT;
  bf16_t* bWtV = bWtK + MAT;
  bf16_t* H1   = bWtV + MAT;                          // h_r [B,S,N]
  bf16_t* H2   = H1 + MAT;                            // h_v
  // buffer reuse after their producers are consumed:
  bf16_t* Qb   = bWr;    // Q  [B,S,D]   (bWr dead after H1)
  bf16_t* Kb   = bWv;    // K            (bWv dead after H2)
  bf16_t* Vtb  = bWtQ;   // V^T [B,D,S]  (bWtQ dead after Q)
  bf16_t* Ob   = bWtK;   // attn out     (bWtK dead after K)

  // 1) convert x, W_O to bf16
  cvt_bf16<<<(unsigned)((MAT + 255) / 256), 256, 0, stream>>>(x, xb, MAT);
  cvt_bf16<<<(unsigned)(((size_t)DD * DD + 255) / 256), 256, 0, stream>>>(
      W_O, wob, (size_t)DD * DD);

  // 2) build circuit weights (rel/val transposed for NT GEMMs)
  dim3 gW(BB * NN_);
  build_w<<<gW, 256, 0, stream>>>(tab_r,   i_r,   n_r,   w_r,   bWr,  0);
  build_w<<<gW, 256, 0, stream>>>(tab_v,   i_v,   n_v,   w_v,   bWv,  0);
  build_w<<<gW, 256, 0, stream>>>(tab_rel, i_rq,  n_rq,  w_rq,  bWtQ, 1);
  build_w<<<gW, 256, 0, stream>>>(tab_rel, i_rk,  n_rk,  w_rk,  bWtK, 1);
  build_w<<<gW, 256, 0, stream>>>(tab_val, i_val, n_val, w_val, bWtV, 1);

  const size_t SB = (size_t)SS * DD;  // per-batch matrix elements
  dim3 g8(SS / 128, NN_ / 128, BB);

  // 3) h_r = x * W_r^T ; h_v = x * W_v^T
  gemm_nt<false, false><<<g8, 256, 0, stream>>>(xb, SB, DD, bWr, SB, DD,
                                                (void*)H1, SB, NN_, DD);
  gemm_nt<false, false><<<g8, 256, 0, stream>>>(xb, SB, DD, bWv, SB, DD,
                                                (void*)H2, SB, NN_, DD);

  // 4) Q = h_r * WtQ^T ; K = h_r * WtK^T ; V^T via transposed epilogue
  gemm_nt<false, false><<<g8, 256, 0, stream>>>(H1, SB, NN_, bWtQ, SB, NN_,
                                                (void*)Qb, SB, DD, NN_);
  gemm_nt<false, false><<<g8, 256, 0, stream>>>(H1, SB, NN_, bWtK, SB, NN_,
                                                (void*)Kb, SB, DD, NN_);
  gemm_nt<true, false><<<g8, 256, 0, stream>>>(H2, SB, NN_, bWtV, SB, NN_,
                                               (void*)Vtb, SB, SS, NN_);

  // 5) causal flash attention
  flash_attn<<<BB * HH * (SS / 64), 128, 0, stream>>>(Qb, Kb, Vtb, Ob);

  // 6) out = attn * W_O^T  (single GEMM, M = B*S, f32 output)
  dim3 gO((BB * SS) / 128, DD / 128, 1);
  gemm_nt<false, true><<<gO, 256, 0, stream>>>(Ob, 0, DD, wob, 0, DD,
                                               d_out, 0, DD, DD);
}
